// _SolarOpenAttention_35691178230202
// MI455X (gfx1250) — matrix operand order repaired and checked by reading, not compiled, not run on hardware
//
#include <hip/hip_runtime.h>
#include <hip/hip_bf16.h>
#include <math.h>

// ---------------------------------------------------------------------------
// Problem constants (from reference)
// ---------------------------------------------------------------------------
constexpr int S_LEN  = 2048;
constexpr int HID    = 4096;
constexpr int D_HEAD = 128;
constexpr int N_H    = 32;
constexpr int N_KV   = 8;
constexpr int GQA_G  = N_H / N_KV;              // 4
constexpr float ROPE_BASE = 10000.0f;

// CDNA5 WMMA vector types (probe-verified signatures)
typedef __attribute__((ext_vector_type(16))) __bf16 v16bf;
typedef __attribute__((ext_vector_type(8)))  __bf16 v8bf;
typedef __attribute__((ext_vector_type(8)))  float  v8f;

static __device__ __forceinline__ v8f wmma_bf16(const v16bf& a, const v16bf& b, const v8f& c) {
    // 8 args: (neg_a, A, neg_b, B, c_mod, C, reuse_a, reuse_b)
    return __builtin_amdgcn_wmma_f32_16x16x32_bf16(false, a, false, b, (short)0, c, false, false);
}

// Two 16B-aligned LDS reads (ds_load_b128) -> one 16-element bf16 fragment.
static __device__ __forceinline__ v16bf ld_frag(const __bf16* p0, const __bf16* p1) {
    v8bf lo = *(const v8bf*)p0;
    v8bf hi = *(const v8bf*)p1;
    return __builtin_shufflevector(lo, hi, 0,1,2,3,4,5,6,7,8,9,10,11,12,13,14,15);
}

// CDNA5 async global->LDS copy (ASYNCcnt, no VGPR round trip).
static __device__ __forceinline__ void async_ld_b128(unsigned lds_off, const void* g) {
    asm volatile("global_load_async_to_lds_b128 %0, %1, off"
                 :: "v"(lds_off), "v"((unsigned long long)(uintptr_t)g)
                 : "memory");
}
static __device__ __forceinline__ void wait_async0() {
    asm volatile("s_wait_asynccnt 0x0" ::: "memory");
}

// ---------------------------------------------------------------------------
// GEMM  C[M,N] = A[M,K] * B[K,N], bf16 WMMA, f32 accumulate.
//  A_BF16:  A is bf16 row-major [M][K]   -> async byte-copy staging
//     else  A is fp32 row-major          -> load+convert staging
//  BT_BF16: B is bf16 PRE-TRANSPOSED [N][K] -> async byte-copy staging
//     else  B is fp32 [K][N]             -> column-read convert staging
//  C_BF16:  write bf16, else fp32
// Block tile 128x128, 256 threads = 8 waves (4x2), wave tile 32x64.
// Double-buffered LDS, ONE barrier per K-step: asyncs for step i+1 are
// issued right after the barrier and overlap with step i's WMMAs.
// ---------------------------------------------------------------------------
constexpr int TM  = 128;
constexpr int TN  = 128;
constexpr int TKC = 32;
constexpr int KP  = TKC + 8;     // 40 elems = 80B row stride (multiple of 16B)

template<bool A_BF16, bool BT_BF16, bool C_BF16>
__global__ __launch_bounds__(256)
void gemm_wmma(const void* __restrict__ Ap, const void* __restrict__ Bp,
               void* __restrict__ Cp, int M, int N, int K)
{
    __shared__ __align__(16) __bf16 As [2][TM][KP];   // [buf][row][k]
    __shared__ __align__(16) __bf16 BsT[2][TN][KP];   // [buf][col][k]

    const int tid   = threadIdx.x;
    const int wave  = tid >> 5;
    const int lane  = tid & 31;
    const int l     = lane & 15;
    const int lhalf = lane >> 4;

    const int wm = wave & 3;              // wave row group -> 32 rows
    const int wn = wave >> 2;             // wave col group -> 64 cols

    const int bm = blockIdx.y * TM;
    const int bn = blockIdx.x * TN;

    const unsigned asB = (unsigned)(uintptr_t)&As[0][0][0];
    const unsigned bsB = (unsigned)(uintptr_t)&BsT[0][0][0];
    constexpr unsigned BUFB = TM * KP * 2;           // bytes per buffer

    const float*  aF = A_BF16  ? nullptr : (const float*)Ap;
    const __bf16* aB = A_BF16  ? (const __bf16*)Ap : nullptr;
    const float*  bF = BT_BF16 ? nullptr : (const float*)Bp;
    const __bf16* bB = BT_BF16 ? (const __bf16*)Bp : nullptr;

    // fp32 staging geometry
    const int ar = tid >> 3, ac = (tid & 7) * 4;          // A: 4 rows/thread
    const int bcol = tid & (TN - 1), bkg = (tid >> 7) * 16; // B: 1 col/half

    auto stage = [&](int buf, int k0) {
        if constexpr (A_BF16) {
            for (int g = 0; g < 2; ++g) {               // 2 x 16B chunks
                const int id = tid + 256 * g;
                const int r = id >> 2, c8 = (id & 3) * 8;
                async_ld_b128(asB + buf * BUFB + (r * KP + c8) * 2,
                              aB + (size_t)(bm + r) * K + k0 + c8);
            }
        } else {
            for (int g = 0; g < 4; ++g) {
                const float4 f = *(const float4*)(aF + (size_t)(bm + ar + 32 * g) * K + k0 + ac);
                __bf16* d = &As[buf][ar + 32 * g][ac];
                d[0] = (__bf16)f.x;  d[1] = (__bf16)f.y;
                d[2] = (__bf16)f.z;  d[3] = (__bf16)f.w;
            }
        }
        if constexpr (BT_BF16) {
            for (int g = 0; g < 2; ++g) {
                const int id = tid + 256 * g;
                const int c = id >> 2, c8 = (id & 3) * 8;
                async_ld_b128(bsB + buf * BUFB + (c * KP + c8) * 2,
                              bB + (size_t)(bn + c) * K + k0 + c8);
            }
        } else {
            const float* bp = bF + (size_t)(k0 + bkg) * N + bn + bcol;
            v8bf w0, w1;
            for (int j = 0; j < 8; ++j) w0[j] = (__bf16)bp[(size_t)j * N];
            for (int j = 0; j < 8; ++j) w1[j] = (__bf16)bp[(size_t)(j + 8) * N];
            *(v8bf*)&BsT[buf][bcol][bkg]     = w0;      // ds_store_b128
            *(v8bf*)&BsT[buf][bcol][bkg + 8] = w1;
        }
    };

    v8f acc[2][4] = {};
    const int nsteps = K / TKC;
    stage(0, 0);

    for (int i = 0; i < nsteps; ++i) {
        wait_async0();              // own asyncs for step i done
        __syncthreads();            // everyone's step i staged; step i-1 compute done
        if (i + 1 < nsteps) stage((i + 1) & 1, (i + 1) * TKC);   // overlaps compute

        const int buf = i & 1;
        v16bf af[2];
        for (int t = 0; t < 2; ++t) {
            const __bf16* row = &As[buf][wm * 32 + t * 16 + l][0];
            af[t] = ld_frag(row + 8 * lhalf, row + 16 + 8 * lhalf);
        }
        v16bf bfv[4];
        for (int t = 0; t < 4; ++t) {
            const __bf16* col = &BsT[buf][wn * 64 + t * 16 + l][0];
            bfv[t] = ld_frag(col + 8 * lhalf, col + 16 + 8 * lhalf);
        }
        for (int ii = 0; ii < 2; ++ii)
            for (int j = 0; j < 4; ++j)
                acc[ii][j] = wmma_bf16(af[ii], bfv[j], acc[ii][j]);
    }

    // C layout: VGPR r -> row base+r (lanes 0-15) / base+r+8 (lanes 16-31)
    for (int i = 0; i < 2; ++i)
        for (int j = 0; j < 4; ++j)
            for (int r = 0; r < 8; ++r) {
                const int row = bm + wm * 32 + i * 16 + r + 8 * lhalf;
                const int col = bn + wn * 64 + j * 16 + l;
                if constexpr (C_BF16)
                    ((__bf16*)Cp)[(size_t)row * N + col] = (__bf16)acc[i][j][r];
                else
                    ((float*)Cp)[(size_t)row * N + col] = acc[i][j][r];
            }
}

// ---------------------------------------------------------------------------
// One-time pre-pack kernels (full path): fp32 -> bf16.
// ---------------------------------------------------------------------------
__global__ __launch_bounds__(256)
void convert_bf16(const float* __restrict__ src, __bf16* __restrict__ dst, long n4)
{
    const long i = ((long)blockIdx.x * blockDim.x + threadIdx.x) * 4;
    if (i >= n4 * 4) return;
    const float4 f = *(const float4*)(src + i);
    dst[i + 0] = (__bf16)f.x;  dst[i + 1] = (__bf16)f.y;
    dst[i + 2] = (__bf16)f.z;  dst[i + 3] = (__bf16)f.w;
}

// W [K][N] fp32 -> WT [N][K] bf16 via 32x32 LDS tile (both sides coalesced)
__global__ __launch_bounds__(256)
void transpose_convert_bf16(const float* __restrict__ W, __bf16* __restrict__ WT,
                            int K, int N)
{
    __shared__ float t[32][33];
    const int tx = threadIdx.x & 31, ty = threadIdx.x >> 5;   // 32 x 8
    const int n0 = blockIdx.x * 32,  k0 = blockIdx.y * 32;
    for (int j = 0; j < 4; ++j)
        t[ty + 8 * j][tx] = W[(size_t)(k0 + ty + 8 * j) * N + n0 + tx];
    __syncthreads();
    for (int j = 0; j < 4; ++j)
        WT[(size_t)(n0 + ty + 8 * j) * K + k0 + tx] = (__bf16)t[tx][ty + 8 * j];
}

// ---------------------------------------------------------------------------
// RoPE in place on bf16 q/k + build transposed V: vbf [S][N_KV*D] -> vT [N_KV][D][S]
// ---------------------------------------------------------------------------
__global__ __launch_bounds__(256)
void rope_vT(__bf16* __restrict__ qbf, __bf16* __restrict__ kbf,
             const __bf16* __restrict__ vbf, __bf16* __restrict__ vT,
             const int* __restrict__ pos)
{
    const int half = D_HEAD / 2;                      // 64
    const long idx = (long)blockIdx.x * blockDim.x + threadIdx.x;
    const long nq = (long)S_LEN * N_H  * half;
    const long nk = (long)S_LEN * N_KV * half;
    const long nv = nk;
    if (idx >= nq + nk + nv) return;

    if (idx < nq + nk) {
        const bool isq = idx < nq;
        const long j   = isq ? idx : idx - nq;
        const int  nh  = isq ? N_H : N_KV;
        const int  s   = (int)(j / (nh * half));
        const int  rem = (int)(j % (nh * half));
        const int  h   = rem / half;
        const int  i   = rem % half;
        __bf16* base = (isq ? qbf : kbf) + (size_t)s * (nh * D_HEAD) + h * D_HEAD;
        const float p   = (float)pos[s];
        const float inv = __powf(ROPE_BASE, -(2.0f * (float)i) / (float)D_HEAD);
        const float ang = p * inv;
        const float c = __cosf(ang), sn = __sinf(ang);
        const float x1 = (float)base[i], x2 = (float)base[i + half];
        base[i]        = (__bf16)(x1 * c - x2 * sn);
        base[i + half] = (__bf16)(x2 * c + x1 * sn);
    } else {
        const long j  = idx - nq - nk;
        const int  s  = (int)(j / (N_KV * half));
        const int  rem = (int)(j % (N_KV * half));
        const int  h  = rem / half;
        const int  i  = rem % half;
        const __bf16* src = vbf + (size_t)s * (N_KV * D_HEAD) + h * D_HEAD;
        vT[((size_t)h * D_HEAD + i)        * S_LEN + s] = src[i];
        vT[((size_t)h * D_HEAD + i + half) * S_LEN + s] = src[i + half];
    }
}

// ---------------------------------------------------------------------------
// Flash attention (causal, GQA). Grid: (head, qblk64). Block: 128 thr = 4 waves.
// All staging via GLOBAL_LOAD_ASYNC_TO_LDS_B128 of bf16 data; fragments are
// ds_load_b128 pairs; online softmax via wave32 shfl_xor; ctx written bf16.
// ---------------------------------------------------------------------------
constexpr int DP = D_HEAD + 8;   // 136 elems = 272B stride (multiple of 16B)

__global__ __launch_bounds__(128)
void flash_attn_wmma(const __bf16* __restrict__ q, const __bf16* __restrict__ k,
                     const __bf16* __restrict__ vT, __bf16* __restrict__ ctx)
{
    const int head = blockIdx.x;
    const int kvh  = head / GQA_G;
    const int qblk = blockIdx.y;
    const int wave = threadIdx.x >> 5;
    const int lane = threadIdx.x & 31;
    const int l     = lane & 15;
    const int lhalf = lane >> 4;
    const int tid   = threadIdx.x;

    const int q0 = qblk * 64 + wave * 16;
    const int kv_end = qblk * 64 + 64;           // uniform across block
    const float scale = 0.08838834764831845f;    // 1/sqrt(128)

    __shared__ __align__(16) __bf16 Qs [64][DP];
    __shared__ __align__(16) __bf16 Ks [32][DP];
    __shared__ __align__(16) __bf16 VsT[D_HEAD][KP];
    __shared__ __align__(16) __bf16 Pst[4][16][KP];

    const unsigned qsB = (unsigned)(uintptr_t)&Qs[0][0];
    const unsigned ksB = (unsigned)(uintptr_t)&Ks[0][0];
    const unsigned vsB = (unsigned)(uintptr_t)&VsT[0][0];

    // ---- stage Q block (64 x 128 bf16) via async copies -----------------
    {
        const int qbase = qblk * 64;
        for (int g = 0; g < 8; ++g) {
            const int id = tid + 128 * g;
            const int r  = id >> 4;
            const int c8 = (id & 15) * 8;
            async_ld_b128(qsB + r * (DP * 2) + c8 * 2,
                          q + (size_t)(qbase + r) * HID + head * D_HEAD + c8);
        }
    }
    wait_async0();
    __syncthreads();

    v16bf qf4[4];
    {
        const __bf16* qrow = &Qs[wave * 16 + l][0];
        for (int c = 0; c < 4; ++c)
            qf4[c] = ld_frag(qrow + c * 32 + 8 * lhalf,
                             qrow + c * 32 + 16 + 8 * lhalf);
    }

    v8f   o[8] = {};
    float m_run[8], l_run[8];
    for (int r = 0; r < 8; ++r) { m_run[r] = -1e30f; l_run[r] = 0.0f; }

    for (int kv0 = 0; kv0 < kv_end; kv0 += 32) {
        // ---- async-stage K (row-major) and pre-transposed V -------------
        for (int g = 0; g < 4; ++g) {
            const int id = tid + 128 * g;
            const int r  = id >> 4;
            const int c8 = (id & 15) * 8;
            async_ld_b128(ksB + r * (DP * 2) + c8 * 2,
                          k + (size_t)(kv0 + r) * (N_KV * D_HEAD) + kvh * D_HEAD + c8);
        }
        for (int g = 0; g < 4; ++g) {
            async_ld_b128(vsB + tid * (KP * 2) + g * 16,
                          vT + ((size_t)kvh * D_HEAD + tid) * S_LEN + kv0 + g * 8);
        }
        wait_async0();
        __syncthreads();

        // ---- scores: S = Q * K^T over d=128 -----------------------------
        v8f sc[2];
        for (int nt = 0; nt < 2; ++nt) {
            v8f s = {};
            const __bf16* krow = &Ks[nt * 16 + l][0];
            for (int c = 0; c < 4; ++c) {
                v16bf bk = ld_frag(krow + c * 32 + 8 * lhalf,
                                   krow + c * 32 + 16 + 8 * lhalf);
                s = wmma_bf16(qf4[c], bk, s);
            }
            sc[nt] = s;
        }

        // ---- scale + causal mask ----------------------------------------
        for (int nt = 0; nt < 2; ++nt)
            for (int r = 0; r < 8; ++r) {
                const int qi = q0 + r + 8 * lhalf;
                const int ki = kv0 + nt * 16 + l;
                const float sv = sc[nt][r] * scale;
                sc[nt][r] = (ki <= qi) ? sv : -1e30f;
            }

        // ---- online softmax ---------------------------------------------
        float alpha[8];
        for (int r = 0; r < 8; ++r) {
            float mx = fmaxf(sc[0][r], sc[1][r]);
            for (int off = 8; off >= 1; off >>= 1)
                mx = fmaxf(mx, __shfl_xor(mx, off, 32));
            const float mnew = fmaxf(m_run[r], mx);
            const float a    = __expf(m_run[r] - mnew);
            m_run[r] = mnew;
            const float p0 = __expf(sc[0][r] - mnew);
            const float p1 = __expf(sc[1][r] - mnew);
            sc[0][r] = p0; sc[1][r] = p1;
            float rs = p0 + p1;
            for (int off = 8; off >= 1; off >>= 1)
                rs += __shfl_xor(rs, off, 32);
            l_run[r] = l_run[r] * a + rs;
            alpha[r] = a;
        }
        for (int t = 0; t < 8; ++t)
            for (int r = 0; r < 8; ++r)
                o[t][r] *= alpha[r];

        // ---- P: C-layout -> A-layout via per-wave LDS bounce ------------
        for (int r = 0; r < 8; ++r) {
            Pst[wave][r + 8 * lhalf][l]      = (__bf16)sc[0][r];
            Pst[wave][r + 8 * lhalf][16 + l] = (__bf16)sc[1][r];
        }
        __syncthreads();
        v16bf pf;
        {
            const __bf16* prow = &Pst[wave][l][0];
            pf = ld_frag(prow + 8 * lhalf, prow + 16 + 8 * lhalf);
        }

        // ---- O += P * V -------------------------------------------------
        for (int t = 0; t < 8; ++t) {
            const __bf16* vcol = &VsT[t * 16 + l][0];
            v16bf bv = ld_frag(vcol + 8 * lhalf, vcol + 16 + 8 * lhalf);
            o[t] = wmma_bf16(pf, bv, o[t]);
        }
        __syncthreads();
    }

    // ---- epilogue: normalize & store bf16 ctx ---------------------------
    for (int t = 0; t < 8; ++t)
        for (int r = 0; r < 8; ++r) {
            const float denom = fmaxf(l_run[r], 1e-20f);
            const int row = q0 + r + 8 * lhalf;
            ctx[(size_t)row * HID + head * D_HEAD + t * 16 + l]
                = (__bf16)(o[t][r] / denom);
        }
}

// ---------------------------------------------------------------------------
// Host-side launch
// ---------------------------------------------------------------------------
extern "C" void kernel_launch(void* const* d_in, const int* in_sizes, int n_in,
                              void* d_out, int out_size, void* d_ws, size_t ws_size,
                              hipStream_t stream)
{
    const float* hs  = (const float*)d_in[0];
    const int*   pos = (const int*)  d_in[1];
    const float* Wq  = (const float*)d_in[2];
    const float* Wk  = (const float*)d_in[3];
    const float* Wv  = (const float*)d_in[4];
    const float* Wo  = (const float*)d_in[5];
    float* out = (float*)d_out;

    // workspace layout (bytes):
    //  [0,16M)    qbf     [16,20M) kbf    [20,24M) vbf
    //  [24,28M)   vT      [28,44M) ctx (bf16)
    //  full path extras:
    //  [44,60M)   hsbf    [60,92M) WqT    [92,100M) WkT
    //  [100,108M) WvT     [108,140M) WoT
    char* w = (char*)d_ws;
    const size_t MB = 1ull << 20;
    __bf16* qbf  = (__bf16*)(w +   0 * MB);
    __bf16* kbf  = (__bf16*)(w +  16 * MB);
    __bf16* vbf  = (__bf16*)(w +  20 * MB);
    __bf16* vT   = (__bf16*)(w +  24 * MB);
    __bf16* ctx  = (__bf16*)(w +  28 * MB);
    __bf16* hsbf = (__bf16*)(w +  44 * MB);
    __bf16* WqT  = (__bf16*)(w +  60 * MB);
    __bf16* WkT  = (__bf16*)(w +  92 * MB);
    __bf16* WvT  = (__bf16*)(w + 100 * MB);
    __bf16* WoT  = (__bf16*)(w + 108 * MB);

    const bool full = ws_size >= 140 * MB;   // constant per harness -> deterministic
    const int NKV_D = N_KV * D_HEAD;         // 1024

    if (full) {
        // one-time bf16 pre-pack: hs + all weights (weights transposed [N][K])
        {
            const long n = (long)S_LEN * HID;
            convert_bf16<<<(int)(n / 4 / 256), 256, 0, stream>>>(hs, hsbf, n / 4);
        }
        transpose_convert_bf16<<<dim3(HID   / 32, HID / 32), 256, 0, stream>>>(Wq, WqT, HID, HID);
        transpose_convert_bf16<<<dim3(NKV_D / 32, HID / 32), 256, 0, stream>>>(Wk, WkT, HID, NKV_D);
        transpose_convert_bf16<<<dim3(NKV_D / 32, HID / 32), 256, 0, stream>>>(Wv, WvT, HID, NKV_D);
        transpose_convert_bf16<<<dim3(HID   / 32, HID / 32), 256, 0, stream>>>(Wo, WoT, HID, HID);

        // fully-async bf16 GEMMs (QKV write bf16 directly)
        gemm_wmma<true,true,true><<<dim3(HID   / TN, S_LEN / TM), 256, 0, stream>>>(hsbf, WqT, qbf, S_LEN, HID,   HID);
        gemm_wmma<true,true,true><<<dim3(NKV_D / TN, S_LEN / TM), 256, 0, stream>>>(hsbf, WkT, kbf, S_LEN, NKV_D, HID);
        gemm_wmma<true,true,true><<<dim3(NKV_D / TN, S_LEN / TM), 256, 0, stream>>>(hsbf, WvT, vbf, S_LEN, NKV_D, HID);
    } else {
        // convert-on-the-fly GEMMs
        gemm_wmma<false,false,true><<<dim3(HID   / TN, S_LEN / TM), 256, 0, stream>>>(hs, Wq, qbf, S_LEN, HID,   HID);
        gemm_wmma<false,false,true><<<dim3(NKV_D / TN, S_LEN / TM), 256, 0, stream>>>(hs, Wk, kbf, S_LEN, NKV_D, HID);
        gemm_wmma<false,false,true><<<dim3(NKV_D / TN, S_LEN / TM), 256, 0, stream>>>(hs, Wv, vbf, S_LEN, NKV_D, HID);
    }

    // RoPE in place + transposed V
    {
        const long pairs = (long)S_LEN * (N_H + N_KV + N_KV) * (D_HEAD / 2);
        rope_vT<<<(int)((pairs + 255) / 256), 256, 0, stream>>>(qbf, kbf, vbf, vT, pos);
    }

    // causal GQA flash attention (bf16 in, bf16 ctx out)
    flash_attn_wmma<<<dim3(N_H, S_LEN / 64), 128, 0, stream>>>(qbf, kbf, vT, ctx);

    // output projection -> fp32 d_out
    if (full)
        gemm_wmma<true,true,false><<<dim3(HID / TN, S_LEN / TM), 256, 0, stream>>>(ctx, WoT, out, S_LEN, HID, HID);
    else
        gemm_wmma<true,false,false><<<dim3(HID / TN, S_LEN / TM), 256, 0, stream>>>(ctx, Wo, out, S_LEN, HID, HID);
}
